// EggNet_82102594830787
// MI455X (gfx1250) — compile-verified, matrix operand order checked by reading, not run on hardware
//
#include <hip/hip_runtime.h>

// ---------------------------------------------------------------------------
// EggNet forward on MI455X (gfx1250): bf16 WMMA GEMMs with fused bias+LN+ReLU,
// static-segment softmax aggregation, bf16 activations end-to-end, fp32 out.
// GEMM: 32-row blocks, each wave computes a 32x16 tile (2 WMMAs sharing one B
// fetch) -> half the global B traffic per v_wmma vs 16-row blocking.
// ---------------------------------------------------------------------------

typedef unsigned short u16;
typedef unsigned int   u32;

typedef __bf16 v16bf __attribute__((ext_vector_type(16)));
typedef float  v8f   __attribute__((ext_vector_type(8)));

union BF16x16 { u32 w[8]; v16bf v; };

__device__ __forceinline__ float bf2f(u16 v) {
    u32 u = ((u32)v) << 16;
    float f; __builtin_memcpy(&f, &u, 4); return f;
}
__device__ __forceinline__ u16 f2bf(float f) {
    u32 u; __builtin_memcpy(&u, &f, 4);
    u32 r = (u + 0x7FFFu + ((u >> 16) & 1u)) >> 16;
    return (u16)r;
}
// sum across an 8-lane group of a wave32 (xor masks 4,2,1 stay in-group)
__device__ __forceinline__ float redux8(float v) {
    v += __shfl_xor(v, 4, 32);
    v += __shfl_xor(v, 2, 32);
    v += __shfl_xor(v, 1, 32);
    return v;
}

// ---------------------------------------------------------------------------
// Weight convert: fp32 [K][N] -> bf16 transposed+padded [Npad][K]
// ---------------------------------------------------------------------------
__global__ void k_conv_w(const float* __restrict__ src, u16* __restrict__ dst,
                         int K, int N, int Npad) {
    long total = (long)K * Npad;
    for (long i = (long)blockIdx.x * 256 + threadIdx.x; i < total;
         i += (long)gridDim.x * 256) {
        int c = (int)(i / K);
        int k = (int)(i - (long)c * K);
        dst[(long)c * K + k] = (c < N) ? f2bf(src[(long)k * N + c]) : (u16)0;
    }
}

// ---------------------------------------------------------------------------
// Fused GEMM + bias + LayerNorm + ReLU, bf16 in / bf16 out, f32 accumulate.
// Block = 256 threads (8 wave32) handles 32 rows. A staged in LDS.
// Each wave: 32x16 output tile = 2 accumulators, one shared B fetch/k-step.
// WT is bf16 transposed weights [Npad][K]. Stats over true N columns.
// ---------------------------------------------------------------------------
__global__ void __launch_bounds__(256)
k_gemm_ln(const u16* __restrict__ A, int lda,
          const u16* __restrict__ WT,
          const float* __restrict__ bias, const float* __restrict__ g,
          const float* __restrict__ bt,
          u16* __restrict__ Out, int ldo,
          int K, int N, int Npad) {
    __shared__ u32   ldsA[32 * 160];      // 32 rows x (K<=320)/2 dword pairs
    __shared__ float ldsC[32 * 256];      // 32 rows x (Npad<=256) f32

    const int tid  = threadIdx.x;
    const int wave = tid >> 5;
    const int lane = tid & 31;
    const int kd2  = K >> 1;
    const long row0 = (long)blockIdx.x * 32;

    // stage A panel (32 x K bf16) into LDS as dwords
    for (int i = tid; i < 32 * kd2; i += 256) {
        int r  = i / kd2;
        int kk = i - r * kd2;
        ldsA[r * kd2 + kk] = *((const u32*)(A + (row0 + r) * lda) + kk);
    }
    __syncthreads();

    const int half  = lane >> 4;          // 0: lanes 0-15, 1: lanes 16-31
    const int l15   = lane & 15;
    const int kbase = half ? 8 : 0;
    const int ntiles = Npad >> 4;

    for (int ct = wave; ct < ntiles; ct += 8) {
        v8f acc0 = {0.f, 0.f, 0.f, 0.f, 0.f, 0.f, 0.f, 0.f};
        v8f acc1 = {0.f, 0.f, 0.f, 0.f, 0.f, 0.f, 0.f, 0.f};
        const u16* wcol = WT + (long)(ct * 16 + l15) * K;
        for (int k0 = 0; k0 < K; k0 += 32) {
            BF16x16 a0, a1, b;
#pragma unroll
            for (int p = 0; p < 8; ++p) {
                const int koff = (p < 4) ? (2 * p) : (16 + 2 * (p - 4));
                const int kk   = k0 + kbase + koff;
                a0.w[p] = ldsA[l15 * kd2 + (kk >> 1)];
                a1.w[p] = ldsA[(16 + l15) * kd2 + (kk >> 1)];
                b.w[p]  = *(const u32*)(wcol + kk);
            }
            acc0 = __builtin_amdgcn_wmma_f32_16x16x32_bf16(
                false, a0.v, false, b.v, (short)0, acc0, false, false);
            acc1 = __builtin_amdgcn_wmma_f32_16x16x32_bf16(
                false, a1.v, false, b.v, (short)0, acc1, false, false);
        }
        // C layout: VGPR r -> local row r + 8*half (+16 for acc1), col below
        const int col = ct * 16 + l15;
        const int rb  = half * 8;
#pragma unroll
        for (int r = 0; r < 8; ++r) {
            ldsC[(rb + r) * 256 + col]      = acc0[r];
            ldsC[(16 + rb + r) * 256 + col] = acc1[r];
        }
    }
    __syncthreads();

    // LayerNorm epilogue: 8 threads per row, 32 rows
    const int row = tid >> 3;
    const int sub = tid & 7;
    float s = 0.f;
    for (int c = sub; c < N; c += 8) {
        float v = ldsC[row * 256 + c] + bias[c];
        ldsC[row * 256 + c] = v;
        s += v;
    }
    s = redux8(s);
    const float mean = s / (float)N;
    float q = 0.f;
    for (int c = sub; c < N; c += 8) {
        float d = ldsC[row * 256 + c] - mean;
        q += d * d;
    }
    q = redux8(q);
    const float rstd = rsqrtf(q / (float)N + 1e-5f);
    const long gr = row0 + row;
    for (int c = sub; c < N; c += 8) {
        float y = (ldsC[row * 256 + c] - mean) * rstd * g[c] + bt[c];
        y = fmaxf(y, 0.f);
        Out[gr * ldo + c] = f2bf(y);
    }
}

// ---------------------------------------------------------------------------
// Encoder layer 0: fp32 x[20000x3] @ W[3x256] + b, LN+ReLU -> bf16
// ---------------------------------------------------------------------------
__global__ void __launch_bounds__(256)
k_enc0_ln(const float* __restrict__ x, const float* __restrict__ W,
          const float* __restrict__ b, const float* __restrict__ g,
          const float* __restrict__ bt, u16* __restrict__ out) {
    const int row = threadIdx.x >> 4;
    const int sub = threadIdx.x & 15;
    const long gr = (long)blockIdx.x * 16 + row;
    const float x0 = x[gr * 3 + 0], x1 = x[gr * 3 + 1], x2 = x[gr * 3 + 2];
    float v[16];
    float s = 0.f;
#pragma unroll
    for (int i = 0; i < 16; ++i) {
        const int c = sub + i * 16;
        float t = x0 * W[c] + x1 * W[256 + c] + x2 * W[512 + c] + b[c];
        v[i] = t; s += t;
    }
    s += __shfl_xor(s, 8, 32);
    s += __shfl_xor(s, 4, 32);
    s += __shfl_xor(s, 2, 32);
    s += __shfl_xor(s, 1, 32);
    const float mean = s * (1.f / 256.f);
    float q = 0.f;
#pragma unroll
    for (int i = 0; i < 16; ++i) { float d = v[i] - mean; q += d * d; }
    q += __shfl_xor(q, 8, 32);
    q += __shfl_xor(q, 4, 32);
    q += __shfl_xor(q, 2, 32);
    q += __shfl_xor(q, 1, 32);
    const float rstd = rsqrtf(q * (1.f / 256.f) + 1e-5f);
#pragma unroll
    for (int i = 0; i < 16; ++i) {
        const int c = sub + i * 16;
        float y = (v[i] - mean) * rstd * g[c] + bt[c];
        out[gr * 256 + c] = f2bf(fmaxf(y, 0.f));
    }
}

// ---------------------------------------------------------------------------
// Build edge-MLP input: [x[start], x[end], (e)] -> bf16 [160000 x Kin]
// end[e] == e>>3 (static). eo stride 80 (dword stride 40), e_new = cols 0..63.
// Dword-granular copies (all row bases 4B-aligned).
// ---------------------------------------------------------------------------
__global__ void k_build_edge_in(const int* __restrict__ start,
                                const u32* __restrict__ xb,   // [20000][64] dw
                                const u32* __restrict__ eo,   // [160000][40] dw
                                u32* __restrict__ out, int Kin2) { // dwords/row
    const long e = blockIdx.x;
    const long s = start[e];
    const long n = e >> 3;
    for (int c2 = threadIdx.x; c2 < Kin2; c2 += 256) {
        u32 v;
        if (c2 < 64)       v = xb[s * 64 + c2];
        else if (c2 < 128) v = xb[n * 64 + (c2 - 64)];
        else               v = eo[e * 40 + (c2 - 128)];
        out[e * Kin2 + c2] = v;
    }
}

// ---------------------------------------------------------------------------
// Static-segment softmax over 8 edges/node + weighted aggregation of e_new.
// Block = 256 threads = 4 nodes x 64 cols.
// ---------------------------------------------------------------------------
__global__ void k_softmax_agg(const u16* __restrict__ eo, u16* __restrict__ agg) {
    const int t = threadIdx.x;
    const long n = (long)blockIdx.x * 4 + (t >> 6);
    const int c = t & 63;
    const long base = n * 8;
    float l[8];
    float mx = -3.0e38f;
#pragma unroll
    for (int j = 0; j < 8; ++j) {
        l[j] = bf2f(eo[(base + j) * 80 + 64]);
        mx = fmaxf(mx, l[j]);
    }
    float s = 0.f;
#pragma unroll
    for (int j = 0; j < 8; ++j) { l[j] = __expf(l[j] - mx); s += l[j]; }
    float acc = 0.f;
#pragma unroll
    for (int j = 0; j < 8; ++j) acc += l[j] * bf2f(eo[(base + j) * 80 + c]);
    agg[n * 64 + c] = f2bf(acc / s);
}

// ---------------------------------------------------------------------------
// Node-MLP input: concat(h[128], agg[64]) -> bf16 [20000 x 192], dword copies
// ---------------------------------------------------------------------------
__global__ void k_concat_node(const u32* __restrict__ h,    // [20000][64] dw
                              const u32* __restrict__ agg,  // [20000][32] dw
                              u32* __restrict__ out) {      // [20000][96] dw
    const long n = blockIdx.x;
    const int c2 = threadIdx.x;
    if (c2 < 96)
        out[n * 96 + c2] = (c2 < 64) ? h[n * 64 + c2] : agg[n * 32 + (c2 - 64)];
}

// ---------------------------------------------------------------------------
// Decoder final layer: bf16 h[20000x256] @ fp32 W[256x12] + b, L2-normalize.
// One wave per row; lanes 0..11 own columns. fp32 output.
// ---------------------------------------------------------------------------
__global__ void __launch_bounds__(256)
k_dec_final(const u16* __restrict__ h, const float* __restrict__ W,
            const float* __restrict__ b, float* __restrict__ out) {
    const int wave = threadIdx.x >> 5;
    const int lane = threadIdx.x & 31;
    const long row = (long)blockIdx.x * 8 + wave;
    float val = 0.f;
    if (lane < 12) {
        float acc = b[lane];
        for (int k = 0; k < 256; ++k)
            acc += bf2f(h[row * 256 + k]) * W[k * 12 + lane];
        val = acc;
    }
    float q = val * val;
    q += __shfl_xor(q, 16, 32);
    q += __shfl_xor(q, 8, 32);
    q += __shfl_xor(q, 4, 32);
    q += __shfl_xor(q, 2, 32);
    q += __shfl_xor(q, 1, 32);
    const float den = fmaxf(sqrtf(q), 1e-12f);
    if (lane < 12) out[row * 12 + lane] = val / den;
}

// ---------------------------------------------------------------------------
// Host orchestration
// ---------------------------------------------------------------------------
extern "C" void kernel_launch(void* const* d_in, const int* in_sizes, int n_in,
                              void* d_out, int out_size, void* d_ws, size_t ws_size,
                              hipStream_t stream) {
    (void)in_sizes; (void)n_in; (void)out_size; (void)ws_size;
    const float* x     = (const float*)d_in[0];
    const int*   edges = (const int*)d_in[1];
    float*       out   = (float*)d_out;

    char* wsb = (char*)d_ws;
    size_t off = 0;
    auto alloc = [&](size_t bytes) -> void* {
        off = (off + 255) & ~(size_t)255;
        void* p = wsb + off;
        off += bytes;
        return p;
    };

    // activation buffers (bf16)
    u16* nb0    = (u16*)alloc(20000L * 256 * 2);
    u16* nb1    = (u16*)alloc(20000L * 256 * 2);
    u16* vbuf   = (u16*)alloc(20000L * 128 * 2);
    u16* hcur   = (u16*)alloc(20000L * 128 * 2);
    u16* aggb   = (u16*)alloc(20000L * 64 * 2);
    u16* nodein = (u16*)alloc(20000L * 192 * 2);
    u16* big0   = (u16*)alloc(160000L * 320 * 2);
    u16* big1   = (u16*)alloc(160000L * 256 * 2);
    u16* eob    = (u16*)alloc(160000L * 80 * 2);

    struct L { u16* wt; const float *b, *g, *bt; int K, N, Npad; };
    auto mk = [&](int idx, int K, int N, int Npad) -> L {
        u16* w = (u16*)alloc((size_t)K * Npad * 2);
        long total = (long)K * Npad;
        int blocks = (int)((total + 255) / 256);
        if (blocks > 4096) blocks = 4096;
        k_conv_w<<<blocks, 256, 0, stream>>>((const float*)d_in[idx], w, K, N, Npad);
        return L{w, (const float*)d_in[idx + 1], (const float*)d_in[idx + 2],
                 (const float*)d_in[idx + 3], K, N, Npad};
    };

    // input index layout (insertion order): x, edges, then per-layer (W,b,g,bt)
    L encL1 = mk(6, 256, 256, 256);
    L encL2 = mk(10, 256, 128, 128);
    L n0L0  = mk(14, 128, 256, 256);
    L n0L1  = mk(18, 256, 128, 128);
    L eL0[4], eL1[4], eL2[4], nL0[4], nL1[4], nL2[4];
    for (int t = 0; t < 4; ++t) {
        int eb = 22 + 12 * t;
        eL0[t] = mk(eb, (t & 1) ? 320 : 256, 256, 256);
        eL1[t] = mk(eb + 4, 256, 256, 256);
        eL2[t] = mk(eb + 8, 256, 65, 80);
        int nbi = 70 + 12 * t;
        nL0[t] = mk(nbi, 192, 256, 256);
        nL1[t] = mk(nbi + 4, 256, 256, 256);
        nL2[t] = mk(nbi + 8, 256, 128, 128);
    }
    L dL0[3], dL1[3];
    const float *dW2[3], *db2[3];
    for (int i = 0; i < 3; ++i) {
        int db = 118 + 12 * i;
        dL0[i] = mk(db, 128, 256, 256);
        dL1[i] = mk(db + 4, 256, 256, 256);
        dW2[i] = (const float*)d_in[db + 8];
        db2[i] = (const float*)d_in[db + 9];
    }

    // M is rows; gemm blocks cover 32 rows each (20000/32=625, 160000/32=5000)
    auto gemm = [&](const u16* A, int lda, const L& l, u16* O, int ldo, int Mb) {
        k_gemm_ln<<<Mb, 256, 0, stream>>>(A, lda, l.wt, l.b, l.g, l.bt,
                                          O, ldo, l.K, l.N, l.Npad);
    };

    // ---- encoder ----
    k_enc0_ln<<<1250, 256, 0, stream>>>(x, (const float*)d_in[2],
                                        (const float*)d_in[3],
                                        (const float*)d_in[4],
                                        (const float*)d_in[5], nb0);
    gemm(nb0, 256, encL1, nb1, 256, 625);
    gemm(nb1, 256, encL2, vbuf, 128, 625);
    // ---- node0 ----
    gemm(vbuf, 128, n0L0, nb0, 256, 625);
    gemm(nb0, 256, n0L1, hcur, 128, 625);

    for (int i = 0; i < 2; ++i) {
        // decoder i on current h
        gemm(hcur, 128, dL0[i], nb0, 256, 625);
        gemm(nb0, 256, dL1[i], nb1, 256, 625);
        k_dec_final<<<2500, 256, 0, stream>>>(nb1, dW2[i], db2[i],
                                              out + (long)i * 240000);
        // GNN block: h resets to encoder output v
        const int* start = edges + (long)i * 320000;  // edges[i,0,:]
        const u16* hin = vbuf;
        for (int j = 0; j < 2; ++j) {
            const int t = 2 * i + j;
            const int Kin = j ? 320 : 256;
            k_build_edge_in<<<160000, 256, 0, stream>>>(
                start, (const u32*)hin, j ? (const u32*)eob : (const u32*)nullptr,
                (u32*)big0, Kin / 2);
            gemm(big0, Kin, eL0[t], big1, 256, 5000);
            gemm(big1, 256, eL1[t], big0, 256, 5000);   // reuse big0, stride 256
            gemm(big0, 256, eL2[t], eob, 80, 5000);     // N=65, Npad=80
            k_softmax_agg<<<5000, 256, 0, stream>>>(eob, aggb);
            k_concat_node<<<20000, 256, 0, stream>>>(
                (const u32*)hin, (const u32*)aggb, (u32*)nodein);
            gemm(nodein, 192, nL0[t], nb0, 256, 625);
            gemm(nb0, 256, nL1[t], nb1, 256, 625);
            gemm(nb1, 256, nL2[t], hcur, 128, 625);
            hin = hcur;
        }
    }
    // final decoder
    gemm(hcur, 128, dL0[2], nb0, 256, 625);
    gemm(nb0, 256, dL1[2], nb1, 256, 625);
    k_dec_final<<<2500, 256, 0, stream>>>(nb1, dW2[2], db2[2], out + 480000L);
}
